// GATLayer_71889162600567
// MI455X (gfx1250) — matrix operand order repaired and checked by reading, not compiled, not run on hardware
//
#include <hip/hip_runtime.h>
#include <math.h>

typedef __attribute__((ext_vector_type(16))) _Float16 v16h;
typedef __attribute__((ext_vector_type(8)))  float    v8f;

#define NNODES 50000
#define EDGE_BLOCKS 2048

// workspace layout (floats):
// [0,256)              M  = u (x) alphaC   (16x16 bilinear-form matrix)
// [256,512)            Ci (16x16 attention-pooled correlation)
// [512,512+N)          logits
// [512+N,512+2N)       att

// ---------------- prep: u = w @ Wc ; M[i*16+j] = u[i]*alphaC[j] ----------------
__global__ void k_prep(const float* __restrict__ w, const float* __restrict__ Wc,
                       const float* __restrict__ alphaC, float* __restrict__ Mout) {
    __shared__ float u[16];
    int t = threadIdx.x;                       // blockDim = 256
    if (t < 16) {
        float s = 0.f;
        #pragma unroll
        for (int o = 0; o < 64; ++o) s += w[o] * Wc[o * 16 + t];
        u[t] = s;
    }
    __syncthreads();
    Mout[t] = u[t >> 4] * alphaC[t & 15];
}

__global__ void k_zero(float* __restrict__ p, int n) {
    int i = blockIdx.x * blockDim.x + threadIdx.x;
    if (i < n) p[i] = 0.f;
}

// -------- pass 1: logits[dst[e]] += S[e] * <C_e, M>  (wave-per-edge streaming) --------
__global__ void k_edge_logits(const float* __restrict__ C, const float* __restrict__ S,
                              const int* __restrict__ dst, const float* __restrict__ M,
                              float* __restrict__ logits, int E) {
    const int lane = threadIdx.x & 31;
    const int wave = (int)((blockIdx.x * blockDim.x + threadIdx.x) >> 5);
    const int nw   = (int)((gridDim.x * blockDim.x) >> 5);

    float m[8];
    #pragma unroll
    for (int k = 0; k < 8; ++k) m[k] = M[lane * 8 + k];

    const float4* Cb = (const float4*)C;
    for (int e = wave; e < E; e += nw) {
        size_t base = (size_t)e * 64 + (size_t)lane * 2;
        if (e + nw < E)
            __builtin_prefetch((const void*)(Cb + base + (size_t)nw * 64), 0, 0);
        float4 a = Cb[base];
        float4 b = Cb[base + 1];
        float d = a.x * m[0] + a.y * m[1] + a.z * m[2] + a.w * m[3]
                + b.x * m[4] + b.y * m[5] + b.z * m[6] + b.w * m[7];
        #pragma unroll
        for (int off = 16; off > 0; off >>= 1) d += __shfl_down(d, off, 32);
        if (lane == 0) atomicAdd(&logits[dst[e]], S[e] * d);
    }
}

// ---------------- softmax over N logits (single workgroup, 3 phases) ----------------
__global__ void k_softmax(const float* __restrict__ logits, float* __restrict__ att, int n) {
    __shared__ float red[1024];
    __shared__ float s_max, s_inv;
    int t = threadIdx.x;
    float mx = -3.4e38f;
    for (int i = t; i < n; i += 1024) mx = fmaxf(mx, logits[i]);
    red[t] = mx; __syncthreads();
    for (int o = 512; o > 0; o >>= 1) {
        if (t < o) red[t] = fmaxf(red[t], red[t + o]);
        __syncthreads();
    }
    if (t == 0) s_max = red[0];
    __syncthreads();
    float sum = 0.f;
    for (int i = t; i < n; i += 1024) sum += __expf(logits[i] - s_max);
    red[t] = sum; __syncthreads();
    for (int o = 512; o > 0; o >>= 1) {
        if (t < o) red[t] += red[t + o];
        __syncthreads();
    }
    if (t == 0) s_inv = 1.f / red[0];
    __syncthreads();
    for (int i = t; i < n; i += 1024) att[i] = __expf(logits[i] - s_max) * s_inv;
}

// -------- pass 2: Ci += att[dst[e]] * S[e] * C_e  (register accumulate, LDS reduce) --------
__global__ void k_edge_ci(const float* __restrict__ C, const float* __restrict__ S,
                          const int* __restrict__ dst, const float* __restrict__ att,
                          float* __restrict__ Ci, int E) {
    __shared__ float red[256];
    int t = threadIdx.x;                        // blockDim = 256
    const int lane = t & 31;
    red[t] = 0.f;
    __syncthreads();

    const int wave = (int)((blockIdx.x * blockDim.x + t) >> 5);
    const int nw   = (int)((gridDim.x * blockDim.x) >> 5);

    float acc[8];
    #pragma unroll
    for (int k = 0; k < 8; ++k) acc[k] = 0.f;

    const float4* Cb = (const float4*)C;
    for (int e = wave; e < E; e += nw) {
        float sc = 0.f;
        if (lane == 0) sc = att[dst[e]] * S[e];
        sc = __shfl(sc, 0, 32);
        size_t base = (size_t)e * 64 + (size_t)lane * 2;
        if (e + nw < E)
            __builtin_prefetch((const void*)(Cb + base + (size_t)nw * 64), 0, 0);
        float4 a = Cb[base];
        float4 b = Cb[base + 1];
        acc[0] += sc * a.x; acc[1] += sc * a.y; acc[2] += sc * a.z; acc[3] += sc * a.w;
        acc[4] += sc * b.x; acc[5] += sc * b.y; acc[6] += sc * b.z; acc[7] += sc * b.w;
    }
    #pragma unroll
    for (int k = 0; k < 8; ++k) atomicAdd(&red[lane * 8 + k], acc[k]);   // ds_add_f32
    __syncthreads();
    atomicAdd(&Ci[t], red[t]);
}

// ---------------- tail: H = Ci @ W1^T via WMMA; sigmoid/relu; Fi2; P ----------------
// 512 threads = 16 waves; wave wv computes the 16x16 output tile for FDIM cols [16wv,16wv+16).
__global__ void k_tail(const float* __restrict__ Ci, const float* __restrict__ W1,
                       const float* __restrict__ b1, const float* __restrict__ W2,
                       const float* __restrict__ b2, const float* __restrict__ Wf,
                       const float* __restrict__ bias, float* __restrict__ out) {
    __shared__ float fi2[256];
    int t = threadIdx.x;
    int wv = t >> 5, lane = t & 31;
    int hf = lane >> 4;            // K-half select
    int m  = lane & 15;            // A row / B column within tile
    int f  = wv * 16 + m;          // global FDIM column

    // A = Ci (16x16 f32 -> f16), K padded 16->32 with zeros.
    // lanes 0-15: h[0..7]=K0..7 ; lanes 16-31: h[0..7]=K8..15 ; h[8..15]=K16..31 = 0
    v16h A, B;
    #pragma unroll
    for (int k = 0; k < 8; ++k) A[k] = (_Float16)Ci[m * 16 + hf * 8 + k];
    #pragma unroll
    for (int k = 8; k < 16; ++k) A[k] = (_Float16)0.0f;
    // B = W1^T tile (32x16): lanes 0-15 h[0..7]=K0..7 ; lanes 16-31 h[0..7]=K8..15 of column f; h[8..15] zero
    #pragma unroll
    for (int k = 0; k < 16; ++k)
        B[k] = (k < 8) ? (_Float16)W1[f * 16 + hf * 8 + k] : (_Float16)0.0f;

    v8f c = {};
    c = __builtin_amdgcn_wmma_f32_16x16x32_f16(false, A, false, B, (short)0, c, false, false);

    // D layout: c[r] -> (M = r + 8*hf, N = m).  Fi2[f] = relu(b2 + sum_i sig(relu(H[i,f]))*W2[i])
    float b2v = b2[0];
    float s = 0.f;
    #pragma unroll
    for (int r = 0; r < 8; ++r) {
        float h = fmaxf(c[r] + b1[f], 0.f);
        float sg = 1.f / (1.f + __expf(-h));
        s += sg * W2[r + hf * 8];
    }
    s += __shfl_xor(s, 16, 32);               // combine the two row-halves of column f
    if (hf == 0) fi2[f] = fmaxf(s + b2v, 0.f);
    __syncthreads();

    if (t < 64) {                              // P = Wf @ Fi2 + bias  -> [1,64]
        float acc = bias[t];
        #pragma unroll 8
        for (int ff = 0; ff < 256; ++ff) acc += Wf[t * 256 + ff] * fi2[ff];
        out[t] = acc;
    }
}

extern "C" void kernel_launch(void* const* d_in, const int* in_sizes, int n_in,
                              void* d_out, int out_size, void* d_ws, size_t ws_size,
                              hipStream_t stream) {
    const float* S      = (const float*)d_in[0];
    const float* C      = (const float*)d_in[1];
    const int*   dst    = (const int*)  d_in[2];
    const float* Wc     = (const float*)d_in[4];
    const float* alphaC = (const float*)d_in[5];
    // alphaf (6) and b (7) only shift logits by a constant -> cancel in softmax
    const float* w      = (const float*)d_in[8];
    const float* Wf     = (const float*)d_in[9];
    const float* bias   = (const float*)d_in[10];
    const float* W1     = (const float*)d_in[11];
    const float* b1     = (const float*)d_in[12];
    const float* W2     = (const float*)d_in[13];
    const float* b2     = (const float*)d_in[14];

    const int E = in_sizes[0];
    float* ws     = (float*)d_ws;
    float* M      = ws;
    float* Ci     = ws + 256;
    float* logits = ws + 512;
    float* att    = ws + 512 + NNODES;
    float* out    = (float*)d_out;

    k_prep<<<1, 256, 0, stream>>>(w, Wc, alphaC, M);
    int nz = NNODES + 256;                                  // Ci + logits are contiguous
    k_zero<<<(nz + 255) / 256, 256, 0, stream>>>(ws + 256, nz);
    k_edge_logits<<<EDGE_BLOCKS, 256, 0, stream>>>(C, S, dst, M, logits, E);
    k_softmax<<<1, 1024, 0, stream>>>(logits, att, NNODES);
    k_edge_ci<<<EDGE_BLOCKS, 256, 0, stream>>>(C, S, dst, att, Ci, E);
    k_tail<<<1, 512, 0, stream>>>(Ci, W1, b1, W2, b2, Wf, bias, out);
}
